// SpMiddleResNetFHD_63385127355250
// MI455X (gfx1250) — matrix operand order repaired and checked by reading, not compiled, not run on hardware
//
#include <hip/hip_runtime.h>

typedef _Float16 f16;
typedef __attribute__((ext_vector_type(16))) _Float16 v16h;
typedef __attribute__((ext_vector_type(8)))  _Float16 v8h;
typedef __attribute__((ext_vector_type(8)))  float    v8f;

#define BN_EPS 0.001f

// ---------------------------------------------------------------------------
// utility kernels
// ---------------------------------------------------------------------------
__global__ void k_scatter(const float* __restrict__ vf, const int* __restrict__ vc,
                          f16* __restrict__ x, float* __restrict__ mask,
                          int n, int H, int W, int C, int CP) {
  int i = blockIdx.x * blockDim.x + threadIdx.x;
  if (i >= n) return;
  int d = vc[i * 4 + 1], h = vc[i * 4 + 2], w = vc[i * 4 + 3];
  size_t idx = ((size_t)d * H + h) * W + w;
  mask[idx] = 1.f;
  for (int c = 0; c < C; ++c) x[idx * CP + c] = (f16)vf[i * C + c];
}

__global__ void k_mask_count(const float* __restrict__ mask, float* __restrict__ cnt, int n) {
  float s = 0.f;
  for (int i = blockIdx.x * blockDim.x + threadIdx.x; i < n; i += gridDim.x * blockDim.x)
    s += mask[i];
  for (int o = 16; o; o >>= 1) s += __shfl_xor(s, o, 32);
  if ((threadIdx.x & 31) == 0) atomicAdd(cnt, s);
}

__global__ void k_down_mask(const float* __restrict__ mi, float* __restrict__ mo,
                            int ID, int IH, int IW, int OD, int OH, int OW,
                            int sd, int sh, int sw, int pd, int ph, int pw,
                            int kd, int kh, int kw) {
  int o = blockIdx.x * blockDim.x + threadIdx.x;
  int nv = OD * OH * OW;
  if (o >= nv) return;
  int od = o / (OH * OW); int r = o % (OH * OW); int oh = r / OW; int ow = r % OW;
  float s = 0.f;
  for (int tz = 0; tz < kd; ++tz) {
    int id = od * sd - pd + tz; if ((unsigned)id >= (unsigned)ID) continue;
    for (int ty = 0; ty < kh; ++ty) {
      int ih = oh * sh - ph + ty; if ((unsigned)ih >= (unsigned)IH) continue;
      for (int tx = 0; tx < kw; ++tx) {
        int iw = ow * sw - pw + tx; if ((unsigned)iw >= (unsigned)IW) continue;
        s += mi[((size_t)id * IH + ih) * IW + iw];
      }
    }
  }
  mo[o] = (s > 0.f) ? 1.f : 0.f;
}

// repack weights f32 -> f16 WMMA-fragment layout:
//   out[((nt*nch + c)*32 + lane)*16 + j] = W[k, nt*16 + (lane&15)],
//   k = 32c + 16*(lane>=16) + j   (zero-padded over cin>=CINo and k>=taps*CINp)
__global__ void k_repack_w(const float* __restrict__ w, f16* __restrict__ out,
                           int taps, int cino, int cinp, int cout, int nch) {
  int total = (cout / 16) * nch * 512;
  for (int i = blockIdx.x * blockDim.x + threadIdx.x; i < total; i += gridDim.x * blockDim.x) {
    int j = i & 15;
    int lane = (i >> 4) & 31;
    int c = (i >> 9) % nch;
    int nt = i / (nch * 512);
    int k = 32 * c + ((lane >> 4) << 4) + j;
    int tap = k / cinp, cin = k - tap * cinp;
    float v = 0.f;
    if (tap < taps && cin < cino)
      v = w[((size_t)tap * cino + cin) * cout + nt * 16 + (lane & 15)];
    out[i] = (f16)v;
  }
}

// ---------------------------------------------------------------------------
// masked batch-norm (tensor is zero outside mask => plain sums == masked sums)
// ---------------------------------------------------------------------------
__global__ void k_bn_stats(const f16* __restrict__ x, float* __restrict__ stats,
                           int nvox, int C, int shift) {
  extern __shared__ float sm[];  // 2*C
  const int t = threadIdx.x;
  const int c = t & (C - 1);
  const int voff = t >> shift;
  const int rpb = blockDim.x >> shift;
  float s = 0.f, q = 0.f;
  for (int v = blockIdx.x * rpb + voff; v < nvox; v += gridDim.x * rpb) {
    float val = (float)x[(size_t)v * C + c];
    s += val; q += val * val;
  }
  for (int i = t; i < 2 * C; i += blockDim.x) sm[i] = 0.f;
  __syncthreads();
  atomicAdd(&sm[c], s);
  atomicAdd(&sm[C + c], q);
  __syncthreads();
  for (int i = t; i < 2 * C; i += blockDim.x) atomicAdd(&stats[i], sm[i]);
}

__global__ void k_bn_finalize(const float* __restrict__ stats, const float* __restrict__ cnt,
                              const float* __restrict__ g, const float* __restrict__ b,
                              float* __restrict__ ss, int C) {
  int c = blockIdx.x * blockDim.x + threadIdx.x;
  if (c >= C) return;
  float n = fmaxf(cnt[0], 1.f);
  float mean = stats[c] / n;
  float var = fmaxf(stats[C + c] / n - mean * mean, 0.f);
  float s = g[c] * rsqrtf(var + BN_EPS);
  ss[c] = s;
  ss[C + c] = b[c] - mean * s;
}

__global__ void k_bn_apply(f16* __restrict__ x, const float* __restrict__ ss,
                           const float* __restrict__ mask, const f16* __restrict__ resid,
                           int relu, int nvox, int C, int shift) {
  const int t = threadIdx.x;
  const int c = t & (C - 1);
  const int voff = t >> shift;
  const int rpb = blockDim.x >> shift;
  const float sc = ss[c], sb = ss[C + c];
  for (int v = blockIdx.x * rpb + voff; v < nvox; v += gridDim.x * rpb) {
    size_t i = (size_t)v * C + c;
    float val = (float)x[i] * sc + sb;
    val *= mask[v];
    if (resid) val += (float)resid[i];
    if (relu) val = fmaxf(val, 0.f);
    x[i] = (f16)val;
  }
}

// ---------------------------------------------------------------------------
// implicit-GEMM conv via WMMA with N-register-blocking (NB output tiles/wave).
// A fragment = two aligned 16B loads, shared across NB WMMAs;
// B fragment = one aligned 32B load per tile from pre-swizzled weights.
// ---------------------------------------------------------------------------
template <int L2C, int KD, int KH, int KW, int NB>
__global__ __launch_bounds__(128)
void k_conv_wmma(const f16* __restrict__ x, const f16* __restrict__ wfrag,
                 const float* __restrict__ bias, const float* __restrict__ omask,
                 f16* __restrict__ y,
                 int ID, int IH, int IW,
                 int OD, int OH, int OW, int COUT,
                 int sd, int sh, int sw, int pd, int ph, int pw) {
  constexpr int CINp = 1 << L2C;
  constexpr int TAPS = KD * KH * KW;
  constexpr int KTOT = TAPS * CINp;
  constexpr int NCH = (KTOT + 31) / 32;

  const int lane = threadIdx.x & 31;
  const int wave = threadIdx.x >> 5;
  const int mtile = blockIdx.x * 4 + wave;
  const int nvox = OD * OH * OW;
  const int mbase = mtile * 16;
  if (mbase >= nvox) return;          // uniform per wave: EXEC stays full for WMMA
  const int lhi = lane >> 4;
  const int ll = lane & 15;

  const int row = mbase + ll;
  const bool rv = row < nvox;
  const int rr = rv ? row : 0;
  const int od = rr / (OH * OW);
  const int r2 = rr - od * (OH * OW);
  const int oh = r2 / OW;
  const int ow = r2 - oh * OW;
  const int bd = od * sd - pd, bh = oh * sh - ph, bw = ow * sw - pw;

  const int nt0 = blockIdx.y * NB;    // first N tile of this wave
  const f16* wf = wfrag + (((size_t)nt0 * NCH) * 32 + lane) * 16;

  auto load8 = [&](int k) -> v8h {
    v8h z = {};
    int tap = k >> L2C;
    if constexpr ((KTOT & 31) != 0) {
      if (tap >= TAPS) return z;      // zero-padded tail chunk
    }
    int cin = k & (CINp - 1);
    int tz = tap / (KH * KW);
    int tr = tap - tz * (KH * KW);
    int ty = tr / KW;
    int tx = tr - ty * KW;
    int id = bd + tz, ih = bh + ty, iw = bw + tx;
    if (rv && (unsigned)id < (unsigned)ID && (unsigned)ih < (unsigned)IH &&
        (unsigned)iw < (unsigned)IW)
      return *(const v8h*)(x + ((((size_t)id * IH + ih) * IW + iw) << L2C) + cin);
    return z;
  };

  v8f acc[NB];
#pragma unroll
  for (int nt = 0; nt < NB; ++nt) acc[nt] = (v8f){};

#pragma unroll 2
  for (int c = 0; c < NCH; ++c) {
    const int kbase = c * 32 + (lhi << 4);
    v8h a0 = load8(kbase);
    v8h a1 = load8(kbase + 8);
    v16h a;
#pragma unroll
    for (int j = 0; j < 8; ++j) { a[j] = a0[j]; a[8 + j] = a1[j]; }
#pragma unroll
    for (int nt = 0; nt < NB; ++nt) {
      v16h b = *(const v16h*)(wf + ((size_t)nt * NCH + c) * 512);
      acc[nt] = __builtin_amdgcn_wmma_f32_16x16x32_f16(false, a, false, b, (short)0,
                                                       acc[nt], false, false);
    }
  }

#pragma unroll
  for (int nt = 0; nt < NB; ++nt) {
    const int col = ((nt0 + nt) << 4) + ll;
    const float bval = bias ? bias[col] : 0.f;
#pragma unroll
    for (int r = 0; r < 8; ++r) {
      int m = mbase + r + (lhi << 3);
      if (m < nvox) {
        float mv = omask ? omask[m] : 1.f;
        y[(size_t)m * COUT + col] = (f16)((acc[nt][r] + bval) * mv);
      }
    }
  }
}

// out[((c*OD+d)*OH+h)*OW+w] = x[((d*OH+h)*OW+w)*C+c]   (transpose(0,4,1,2,3))
__global__ void k_final(const f16* __restrict__ x, float* __restrict__ out,
                        int OD, int OH, int OW, int C) {
  int total = C * OD * OH * OW;
  int i = blockIdx.x * blockDim.x + threadIdx.x;
  if (i >= total) return;
  int w = i % OW; int t = i / OW;
  int h = t % OH; t /= OH;
  int d = t % OD; int c = t / OD;
  out[i] = (float)x[(((size_t)d * OH + h) * OW + w) * C + c];
}

// ---------------------------------------------------------------------------
// host orchestration
// ---------------------------------------------------------------------------
static inline int gmin_i(long a, long b) { return (int)(a < b ? a : b); }

extern "C" void kernel_launch(void* const* d_in, const int* in_sizes, int n_in,
                              void* d_out, int out_size, void* d_ws, size_t ws_size,
                              hipStream_t stream) {
  (void)out_size; (void)ws_size;
  if (n_in < 81) return;

  // ---- parameter pointer resolution (insertion-order or sorted-flatten) ----
  const float* P[79];
  const float* vf; const int* vc;
  auto fill_sorted = [&](int base) {
    static const int singles[15] = {21, 20, 40, 39, 59, 58, 78, 77, 2, 1, 0, 19, 38, 57, 76};
    int idx = base;
    for (int i = 0; i < 15; i++) P[singles[i]] = (const float*)d_in[idx++];
    static const int blockBase[8] = {3, 11, 22, 30, 41, 49, 60, 68};
    static const int boff[8] = {3, 2, 7, 6, 1, 0, 5, 4};
    for (int b = 0; b < 8; b++)
      for (int j = 0; j < 8; j++) P[blockBase[b] + boff[j]] = (const float*)d_in[idx++];
  };
  if (in_sizes[0] == 600000) {
    vf = (const float*)d_in[0];
    vc = (const int*)d_in[1];
    if (in_sizes[2] == 3 * 3 * 3 * 5 * 16) {
      for (int i = 0; i < 79; i++) P[i] = (const float*)d_in[2 + i];
    } else {
      fill_sorted(2);
    }
  } else {
    fill_sorted(0);
    vc = (const int*)d_in[79];
    vf = (const float*)d_in[80];
  }

  // ---- geometry ----
  const int D0 = 41, H0 = 200, W0 = 200; const int nv0 = D0 * H0 * W0;
  const int D1 = 21, H1 = 100, W1 = 100; const int nv1 = D1 * H1 * W1;
  const int D2 = 11, H2 = 50,  W2 = 50;  const int nv2 = D2 * H2 * W2;
  const int D3 = 5,  H3 = 25,  W3 = 25;  const int nv3 = D3 * H3 * W3;
  const int D4 = 2,  H4 = 25,  W4 = 25;  const int nv4 = D4 * H4 * W4;

  // ---- workspace bump allocator ----
  char* wsb = (char*)d_ws;
  size_t off = 0;
  auto alloc = [&](size_t bytes) -> void* {
    off = (off + 255) & ~(size_t)255;
    void* p = wsb + off;
    off += bytes;
    return p;
  };
  f16* x5 = (f16*)alloc((size_t)nv0 * 8 * 2);         // CIN padded 5 -> 8
  float* mask0 = (float*)alloc((size_t)nv0 * 4);
  float* mask1 = (float*)alloc((size_t)nv1 * 4);
  float* mask2 = (float*)alloc((size_t)nv2 * 4);
  float* mask3 = (float*)alloc((size_t)nv3 * 4);
  float* mask4 = (float*)alloc((size_t)nv4 * 4);
  f16* A0 = (f16*)alloc((size_t)nv0 * 16 * 2); f16* B0 = (f16*)alloc((size_t)nv0 * 16 * 2);
  f16* C0 = (f16*)alloc((size_t)nv0 * 16 * 2);
  f16* A1 = (f16*)alloc((size_t)nv1 * 32 * 2); f16* B1 = (f16*)alloc((size_t)nv1 * 32 * 2);
  f16* C1 = (f16*)alloc((size_t)nv1 * 32 * 2);
  f16* A2 = (f16*)alloc((size_t)nv2 * 64 * 2); f16* B2 = (f16*)alloc((size_t)nv2 * 64 * 2);
  f16* C2 = (f16*)alloc((size_t)nv2 * 64 * 2);
  f16* A3 = (f16*)alloc((size_t)nv3 * 128 * 2); f16* B3 = (f16*)alloc((size_t)nv3 * 128 * 2);
  f16* C3 = (f16*)alloc((size_t)nv3 * 128 * 2);
  f16* A4 = (f16*)alloc((size_t)nv4 * 128 * 2);
  float* stats = (float*)alloc(256 * 4);
  float* ssbuf = (float*)alloc(256 * 4);
  float* cnt = (float*)alloc(8 * 4);

  // ---- repack conv weights into f16 WMMA fragments ----
  struct WInfo { int idx; int taps; int cino; int cinp; int cout; };
  static const WInfo wtab[] = {
      {0, 27, 5, 8, 16},
      {3, 27, 16, 16, 16}, {7, 27, 16, 16, 16}, {11, 27, 16, 16, 16}, {15, 27, 16, 16, 16},
      {19, 27, 16, 16, 32},
      {22, 27, 32, 32, 32}, {26, 27, 32, 32, 32}, {30, 27, 32, 32, 32}, {34, 27, 32, 32, 32},
      {38, 27, 32, 32, 64},
      {41, 27, 64, 64, 64}, {45, 27, 64, 64, 64}, {49, 27, 64, 64, 64}, {53, 27, 64, 64, 64},
      {57, 27, 64, 64, 128},
      {60, 27, 128, 128, 128}, {64, 27, 128, 128, 128}, {68, 27, 128, 128, 128},
      {72, 27, 128, 128, 128},
      {76, 3, 128, 128, 128},
  };
  f16* WF[79] = {nullptr};
  for (const WInfo& wi : wtab) {
    int nch = (wi.taps * wi.cinp + 31) / 32;
    int total = (wi.cout / 16) * nch * 512;
    f16* dst = (f16*)alloc((size_t)total * sizeof(f16));
    WF[wi.idx] = dst;
    k_repack_w<<<gmin_i((total + 255) / 256, 2048), 256, 0, stream>>>(
        P[wi.idx], dst, wi.taps, wi.cino, wi.cinp, wi.cout, nch);
  }

  // ---- helpers ----
  auto conv = [&](const f16* xin, int widx, const float* bias, const float* om, f16* yout,
                  int ID, int IH, int IW, int OD, int OH, int OW, int CO,
                  int sd, int sh, int sw, int pd, int ph, int pw, int cinp, int extra) {
    int nvo = OD * OH * OW;
    int nb = (CO == 16) ? 1 : (CO == 128) ? 4 : 2;
    dim3 grid(((nvo + 15) / 16 + 3) / 4, CO / (16 * nb));
#define LCONV(L2C, KD, KH, KW, NB)                                                 \
    k_conv_wmma<L2C, KD, KH, KW, NB><<<grid, 128, 0, stream>>>(                    \
        xin, WF[widx], bias, om, yout, ID, IH, IW, OD, OH, OW, CO,                 \
        sd, sh, sw, pd, ph, pw)
    if (extra)            { LCONV(7, 3, 1, 1, 4); }          // extra 128->128 (3,1,1)
    else if (cinp == 8)   { LCONV(3, 3, 3, 3, 1); }          // conv_in 8->16
    else if (cinp == 16 && CO == 16) { LCONV(4, 3, 3, 3, 1); } // res0
    else if (cinp == 16)  { LCONV(4, 3, 3, 3, 2); }          // down1 16->32
    else if (cinp == 32 && CO == 32) { LCONV(5, 3, 3, 3, 2); } // res1
    else if (cinp == 32)  { LCONV(5, 3, 3, 3, 2); }          // down2 32->64
    else if (cinp == 64 && CO == 64) { LCONV(6, 3, 3, 3, 2); } // res2
    else if (cinp == 64)  { LCONV(6, 3, 3, 3, 4); }          // down3 64->128
    else                  { LCONV(7, 3, 3, 3, 4); }          // res3
#undef LCONV
  };
  auto bn = [&](f16* xb, int nvo, int C, const float* g, const float* b,
                const float* maskp, float* cntp, const f16* resid, int relu) {
    int shift = (C == 16) ? 4 : (C == 32) ? 5 : (C == 64) ? 6 : 7;
    int rpb = 256 >> shift;
    (void)hipMemsetAsync(stats, 0, 2 * C * sizeof(float), stream);
    k_bn_stats<<<gmin_i((nvo + rpb - 1) / rpb, 1024), 256, 2 * C * sizeof(float), stream>>>(
        xb, stats, nvo, C, shift);
    k_bn_finalize<<<1, C, 0, stream>>>(stats, cntp, g, b, ssbuf, C);
    k_bn_apply<<<gmin_i((nvo + rpb - 1) / rpb, 4096), 256, 0, stream>>>(
        xb, ssbuf, maskp, resid, relu, nvo, C, shift);
  };
  auto run_block = [&](int pbase, f16*& X, f16*& T1, f16*& T2,
                       int Dd, int Hh, int Ww, int Cc, const float* maskp, float* cntp) {
    int nvo = Dd * Hh * Ww;
    conv(X, pbase + 0, P[pbase + 1], maskp, T1, Dd, Hh, Ww, Dd, Hh, Ww, Cc,
         1, 1, 1, 1, 1, 1, Cc, 0);
    bn(T1, nvo, Cc, P[pbase + 2], P[pbase + 3], maskp, cntp, nullptr, 1);
    conv(T1, pbase + 4, P[pbase + 5], maskp, T2, Dd, Hh, Ww, Dd, Hh, Ww, Cc,
         1, 1, 1, 1, 1, 1, Cc, 0);
    bn(T2, nvo, Cc, P[pbase + 6], P[pbase + 7], maskp, cntp, X, 1);
    f16* t = X; X = T2; T2 = t;
  };
  auto run_down = [&](int widx, f16* Xin, int ID, int IH, int IW, int CI,
                      f16* Xout, int OD, int OH, int OW, int CO,
                      const float* mi, float* mo, float* cntp,
                      int sd, int sh, int sw, int pd, int ph, int pw,
                      int kd, int kh, int kw, int extra) {
    int nvo = OD * OH * OW;
    k_down_mask<<<(nvo + 255) / 256, 256, 0, stream>>>(mi, mo, ID, IH, IW, OD, OH, OW,
                                                       sd, sh, sw, pd, ph, pw, kd, kh, kw);
    k_mask_count<<<gmin_i((nvo + 255) / 256, 2048), 256, 0, stream>>>(mo, cntp, nvo);
    conv(Xin, widx, nullptr, mo, Xout, ID, IH, IW, OD, OH, OW, CO,
         sd, sh, sw, pd, ph, pw, CI, extra);
    bn(Xout, nvo, CO, P[widx + 1], P[widx + 2], mo, cntp, nullptr, 1);
  };

  // ---- forward pass ----
  (void)hipMemsetAsync(x5, 0, (size_t)nv0 * 8 * 2, stream);
  (void)hipMemsetAsync(mask0, 0, (size_t)nv0 * 4, stream);
  (void)hipMemsetAsync(cnt, 0, 8 * 4, stream);
  k_scatter<<<(120000 + 255) / 256, 256, 0, stream>>>(vf, vc, x5, mask0, 120000, H0, W0, 5, 8);
  k_mask_count<<<gmin_i((nv0 + 255) / 256, 2048), 256, 0, stream>>>(mask0, &cnt[0], nv0);

  conv(x5, 0, nullptr, mask0, A0, D0, H0, W0, D0, H0, W0, 16,
       1, 1, 1, 1, 1, 1, 8, 0);                               // conv_in 5(8)->16
  bn(A0, nv0, 16, P[1], P[2], mask0, &cnt[0], nullptr, 1);

  f16 *X = A0, *T1 = B0, *T2 = C0;
  run_block(3,  X, T1, T2, D0, H0, W0, 16, mask0, &cnt[0]);   // res0a
  run_block(11, X, T1, T2, D0, H0, W0, 16, mask0, &cnt[0]);   // res0b

  run_down(19, X, D0, H0, W0, 16, A1, D1, H1, W1, 32, mask0, mask1, &cnt[1],
           2, 2, 2, 1, 1, 1, 3, 3, 3, 0);                     // down1 + bn1
  X = A1; T1 = B1; T2 = C1;
  run_block(22, X, T1, T2, D1, H1, W1, 32, mask1, &cnt[1]);   // res1a
  run_block(30, X, T1, T2, D1, H1, W1, 32, mask1, &cnt[1]);   // res1b

  run_down(38, X, D1, H1, W1, 32, A2, D2, H2, W2, 64, mask1, mask2, &cnt[2],
           2, 2, 2, 1, 1, 1, 3, 3, 3, 0);                     // down2 + bn2
  X = A2; T1 = B2; T2 = C2;
  run_block(41, X, T1, T2, D2, H2, W2, 64, mask2, &cnt[2]);   // res2a
  run_block(49, X, T1, T2, D2, H2, W2, 64, mask2, &cnt[2]);   // res2b

  run_down(57, X, D2, H2, W2, 64, A3, D3, H3, W3, 128, mask2, mask3, &cnt[3],
           2, 2, 2, 0, 1, 1, 3, 3, 3, 0);                     // down3 + bn3
  X = A3; T1 = B3; T2 = C3;
  run_block(60, X, T1, T2, D3, H3, W3, 128, mask3, &cnt[3]);  // res3a
  run_block(68, X, T1, T2, D3, H3, W3, 128, mask3, &cnt[3]);  // res3b

  run_down(76, X, D3, H3, W3, 128, A4, D4, H4, W4, 128, mask3, mask4, &cnt[4],
           2, 1, 1, 0, 0, 0, 3, 1, 1, 1);                     // extra + bn4

  int total = 128 * nv4;
  k_final<<<(total + 255) / 256, 256, 0, stream>>>(A4, (float*)d_out, D4, H4, W4, 128);
}
